// MySelfAttention_82240033784451
// MI455X (gfx1250) — matrix operand order repaired and checked by reading, not compiled, not run on hardware
//
#include <hip/hip_runtime.h>
#include <hip/hip_bf16.h>

// MI455X / gfx1250 self-attention: bf16 WMMA (f32 accum) + flash softmax,
// Tensor Data Mover (TDM) double-buffered LDS staging, global prefetch.
//   K0: x f32 -> bf16 (one pass; makes X tiles TDM-able)
//   K1: Q/K/V = x@W + b   (v_wmma_f32_16x16x32_bf16; TDM-staged X tiles)
//   K2: flash attention   (TDM-staged K tiles, prefetched+transposed V tiles)

typedef __bf16 bf16_t;
typedef __attribute__((ext_vector_type(16))) __bf16 v16bf;
typedef __attribute__((ext_vector_type(8)))  float  v8f;
typedef unsigned int u32x4 __attribute__((ext_vector_type(4)));
typedef int          i32x8 __attribute__((ext_vector_type(8)));
typedef int          i32x4 __attribute__((ext_vector_type(4)));

#define DD   1024
#define SEQ  2048
#define NB   8

#if defined(__has_builtin)
#  if __has_builtin(__builtin_amdgcn_tensor_load_to_lds) && \
      __has_builtin(__builtin_amdgcn_s_wait_tensorcnt)
#    define HAS_TDM 1
#  endif
#endif

union AF { v16bf v; unsigned int u[8]; };

static __device__ __forceinline__ v8f vzero8() {
    v8f z;
#pragma unroll
    for (int i = 0; i < 8; ++i) z[i] = 0.0f;
    return z;
}

// 16-bit A-matrix 16x32 fragment K index for VGPR v, lane-group g (ISA 7.12.2)
static __device__ __forceinline__ int kA(int v, int g) {
    return (v < 4 ? 2 * v : 16 + 2 * (v - 4)) + 8 * g;
}
// 16-bit B-matrix 32x16 fragment K index (same order as A fragment)
static __device__ __forceinline__ int kB(int v, int g) {
    return (v < 4 ? 2 * v : 16 + 2 * (v - 4)) + 8 * g;
}

#ifdef HAS_TDM
// 2D bf16 tile load via Tensor Data Mover (D# groups per ISA 08_async_tensor).
// pad_enable=1, pad_amount=1 DWORD every (1<<pad_interval_code) 8-byte units.
// This toolchain's builtin takes 6 args: (g0 u32x4, g1 i32x8, g2 i32x4,
// g3 i32x4, g2b i32x8, cpol i32).
static __device__ __forceinline__ void tdm_load_tile_bf16(
    unsigned lds_addr, const bf16_t* gptr,
    unsigned tile_d0, unsigned tile_d1,
    unsigned tensor_d0, unsigned tensor_d1,
    unsigned stride0_elems, unsigned pad_interval_code)
{
    unsigned long long ga = (unsigned long long)(uintptr_t)gptr;
    u32x4 g0;
    g0[0] = 1u;                                        // count=1, user D#
    g0[1] = lds_addr;                                  // lds_addr (bytes)
    g0[2] = (unsigned)ga;                              // global_addr[31:0]
    g0[3] = (unsigned)((ga >> 32) & 0x01FFFFFFull)     // global_addr[56:32]
          | (2u << 30);                                // type=2 ("image")
    i32x8 g1;
    g1[0] = (int)((1u << 16)                           // data_size = 2 bytes
                | (1u << 20)                           // pad_enable
                | (pad_interval_code << 22));          // pad_amount code 0 = 1 DW
    g1[1] = (int)((tensor_d0 & 0xFFFFu) << 16);        // tensor_dim0[15:0]
    g1[2] = (int)(((tensor_d0 >> 16) & 0xFFFFu) | ((tensor_d1 & 0xFFFFu) << 16));
    g1[3] = (int)(((tensor_d1 >> 16) & 0xFFFFu) | ((tile_d0 & 0xFFFFu) << 16));
    g1[4] = (int)(tile_d1 & 0xFFFFu);                  // tile_dim1, tile_dim2=0
    g1[5] = (int)stride0_elems;                        // tensor_dim0_stride[31:0]
    g1[6] = 0;
    g1[7] = 0;
    i32x4 gz4 = {0, 0, 0, 0};
    i32x8 gz8 = {0, 0, 0, 0, 0, 0, 0, 0};
    __builtin_amdgcn_tensor_load_to_lds(g0, g1, gz4, gz4, gz8, 0);
}
#endif

// ---------------------------------------------------------------------------
// Kernel 0: convert x (f32) -> bf16, 8 elems/thread.
// ---------------------------------------------------------------------------
__global__ __launch_bounds__(256) void cvt_bf16_kernel(
    const float* __restrict__ X, bf16_t* __restrict__ Y)
{
    size_t i = ((size_t)blockIdx.x * 256 + threadIdx.x) * 8;
    float4 a = *(const float4*)&X[i];
    float4 b = *(const float4*)&X[i + 4];
    union { uint4 q; bf16_t h[8]; } cv;
    cv.h[0] = (bf16_t)a.x; cv.h[1] = (bf16_t)a.y;
    cv.h[2] = (bf16_t)a.z; cv.h[3] = (bf16_t)a.w;
    cv.h[4] = (bf16_t)b.x; cv.h[5] = (bf16_t)b.y;
    cv.h[6] = (bf16_t)b.z; cv.h[7] = (bf16_t)b.w;
    *(uint4*)&Y[i] = cv.q;
}

// ---------------------------------------------------------------------------
// Kernel 1: QKV projection.  C[16384,1024] = Xbf @ W + b, bf16 output.
// Block = 256 thr (8 waves, 4x2), tile 128x128, k-step 32.
// X tiles: TDM double-buffered (pad_interval=16DW -> LDS stride 34 bf16).
// ---------------------------------------------------------------------------
#define AS_STRIDE 34
#define BS_STRIDE 34

__global__ __launch_bounds__(256) void qkv_gemm_kernel(
    const bf16_t* __restrict__ Xb,
    const float* __restrict__ Wq, const float* __restrict__ bq,
    const float* __restrict__ Wk, const float* __restrict__ bk,
    const float* __restrict__ Wv, const float* __restrict__ bv,
    bf16_t* __restrict__ Qo, bf16_t* __restrict__ Ko, bf16_t* __restrict__ Vo)
{
    const float* W; const float* bias; bf16_t* O;
    if (blockIdx.z == 0)      { W = Wq; bias = bq; O = Qo; }
    else if (blockIdx.z == 1) { W = Wk; bias = bk; O = Ko; }
    else                      { W = Wv; bias = bv; O = Vo; }

    __shared__ bf16_t As[2][128 * AS_STRIDE];   // [row][k], TDM-padded stride
    __shared__ bf16_t Bs[128 * BS_STRIDE];      // [n][k]  (W tile transposed)

    const int t    = threadIdx.x;
    const int wave = t >> 5;
    const int lane = t & 31;
    const int wm   = wave >> 1;
    const int wn   = wave & 1;
    const int col  = lane & 15;
    const int g    = lane >> 4;
    const int r0   = blockIdx.y * 128;
    const int n0   = blockIdx.x * 128;

    v8f acc[2][4];
#pragma unroll
    for (int a = 0; a < 2; ++a)
#pragma unroll
        for (int b = 0; b < 4; ++b) acc[a][b] = vzero8();

#ifdef HAS_TDM
    if (wave == 0)    // prologue: DMA first X tile into buffer 0
        tdm_load_tile_bf16((unsigned)(uintptr_t)&As[0][0],
                           Xb + (size_t)r0 * DD,
                           32, 128, 1024, 128, 1024, 3);
#endif

    for (int k0 = 0; k0 < DD; k0 += 32) {
        const int cur = (k0 >> 5) & 1;
        __syncthreads();
        // stage W tile transposed + f32->bf16: Bs[n][k]
        for (int idx = t; idx < 32 * 128; idx += 256) {
            int k = idx >> 7, n = idx & 127;
            Bs[n * BS_STRIDE + k] = (bf16_t)W[(size_t)(k0 + k) * DD + n0 + n];
        }
#ifdef HAS_TDM
        if (wave == 0) {
            if (k0 + 32 < DD) {   // DMA next X tile, then wait for current
                tdm_load_tile_bf16((unsigned)(uintptr_t)&As[cur ^ 1][0],
                                   Xb + (size_t)r0 * DD + (k0 + 32),
                                   32, 128, 1024, 128, 1024, 3);
                __builtin_amdgcn_s_wait_tensorcnt(1);
            } else {
                __builtin_amdgcn_s_wait_tensorcnt(0);
            }
        }
#else
        for (int idx = t; idx < 128 * 32; idx += 256) {
            int r = idx >> 5, k = idx & 31;
            As[cur][r * AS_STRIDE + k] = Xb[(size_t)(r0 + r) * DD + k0 + k];
        }
#endif
        __syncthreads();

        AF a[2], b[4];
#pragma unroll
        for (int tm = 0; tm < 2; ++tm) {
            int row = wm * 32 + tm * 16 + col;
#pragma unroll
            for (int v = 0; v < 8; ++v)
                a[tm].u[v] = *(const unsigned int*)&As[cur][row * AS_STRIDE + kA(v, g)];
        }
#pragma unroll
        for (int tn = 0; tn < 4; ++tn) {
            int n = wn * 64 + tn * 16 + col;
#pragma unroll
            for (int v = 0; v < 8; ++v)
                b[tn].u[v] = *(const unsigned int*)&Bs[n * BS_STRIDE + kB(v, g)];
        }
#pragma unroll
        for (int tm = 0; tm < 2; ++tm)
#pragma unroll
            for (int tn = 0; tn < 4; ++tn)
                acc[tm][tn] = __builtin_amdgcn_wmma_f32_16x16x32_bf16(
                    false, a[tm].v, false, b[tn].v, (short)0, acc[tm][tn],
                    false, false);
    }

    // epilogue: + bias, convert to bf16
#pragma unroll
    for (int tm = 0; tm < 2; ++tm) {
#pragma unroll
        for (int tn = 0; tn < 4; ++tn) {
            int gcol = n0 + wn * 64 + tn * 16 + col;
            float bb = bias[gcol];
#pragma unroll
            for (int i = 0; i < 8; ++i) {
                int grow = r0 + wm * 32 + tm * 16 + i + 8 * g;
                O[(size_t)grow * DD + gcol] = (bf16_t)(acc[tm][tn][i] + bb);
            }
        }
    }
}

// ---------------------------------------------------------------------------
// Kernel 2: flash attention. Block = 256 thr (8 waves) handles 16 queries of
// one batch; wave w owns output d-slice [w*128, w*128+128).
// K tiles: TDM double-buffered, pad_interval=256DW -> row stride 1028 bf16
//          (bank stride 2 -> conflict-free B-fragment ds_load_b32).
// ---------------------------------------------------------------------------
#define KS_STRIDE2 1028
#define KBUF_BYTES (32 * KS_STRIDE2 * 2)       // 65792 per buffer
#define VT_OFF     (2 * KBUF_BYTES)            // 131584
#define VT_STRIDE  34
#define VT_BYTES   (1024 * VT_STRIDE * 2)      // 69632
#define SRED_OFF   (VT_OFF + VT_BYTES)         // 201216 : 8*16*32 f32
#define SFIN_OFF   (SRED_OFF + 16384)          // 16*32 f32
#define PS_OFF     (SFIN_OFF + 2048)           // 16*34 bf16
#define PS_STRIDE  34
#define STAT_OFF   (PS_OFF + 1088)             // m/l/alpha
#define SMEM_BYTES (STAT_OFF + 192)            // 220928 (< 320KB WGP LDS)

__global__ __launch_bounds__(256) void flash_attn_kernel(
    const bf16_t* __restrict__ Q, const bf16_t* __restrict__ K,
    const bf16_t* __restrict__ V, float* __restrict__ Out)
{
    extern __shared__ char smem[];
    bf16_t* Vt    = (bf16_t*)(smem + VT_OFF);
    float*  Sred  = (float*)(smem + SRED_OFF);
    float*  Sfin  = (float*)(smem + SFIN_OFF);
    bf16_t* Ps    = (bf16_t*)(smem + PS_OFF);
    float*  row_m = (float*)(smem + STAT_OFF);
    float*  row_l = row_m + 16;
    float*  alpha = row_l + 16;

    const int t    = threadIdx.x;
    const int wave = t >> 5;
    const int lane = t & 31;
    const int col  = lane & 15;
    const int g    = lane >> 4;
    const int padk = (wave >= 4) ? 2 : 0;   // TDM mid-row pad for d >= 512
    const int q0   = blockIdx.x * 16;
    const size_t base = (size_t)blockIdx.y * SEQ * DD;
    const bf16_t* Qb = Q + base;
    const bf16_t* Kb = K + base;
    const bf16_t* Vb = V + base;
    float* Ob = Out + base;

    // load this wave's Q fragments (16 rows x 128 d-slice) once
    AF qf[4];
#pragma unroll
    for (int c = 0; c < 4; ++c)
#pragma unroll
        for (int v = 0; v < 8; ++v)
            qf[c].u[v] = *(const unsigned int*)
                &Qb[(size_t)(q0 + col) * DD + wave * 128 + c * 32 + kA(v, g)];

    v8f of[8];
#pragma unroll
    for (int i = 0; i < 8; ++i) of[i] = vzero8();

    if (t < 16) { row_m[t] = -1.0e30f; row_l[t] = 0.0f; }

#ifdef HAS_TDM
    if (wave == 0)    // prologue: DMA first K tile into buffer 0
        tdm_load_tile_bf16((unsigned)(uintptr_t)smem, Kb,
                           1024, 32, 1024, 32, 1024, 7);
#endif
    __syncthreads();

    for (int kt = 0; kt < SEQ; kt += 32) {
        const int cur = (kt >> 5) & 1;
        bf16_t* Ks = (bf16_t*)(smem + (unsigned)cur * KBUF_BYTES);
#ifdef HAS_TDM
        if (wave == 0) {
            if (kt + 32 < SEQ) {   // DMA next K tile, then wait for current
                tdm_load_tile_bf16(
                    (unsigned)(uintptr_t)(smem + (unsigned)(cur ^ 1) * KBUF_BYTES),
                    Kb + (size_t)(kt + 32) * DD, 1024, 32, 1024, 32, 1024, 7);
                __builtin_amdgcn_s_wait_tensorcnt(1);
            } else {
                __builtin_amdgcn_s_wait_tensorcnt(0);
            }
        }
#else
        for (int vi = t; vi < 32 * 512; vi += 256) {
            int key = vi >> 9;
            int dp  = (vi & 511) * 2;
            int off = key * KS_STRIDE2 + dp + (dp >= 512 ? 2 : 0);
            *(unsigned int*)&Ks[off] =
                *(const unsigned int*)&Kb[(size_t)(kt + key) * DD + dp];
        }
#endif
        __syncthreads();

        // prefetch V tile cache lines while scores compute
        {
            const bf16_t* V0 = Vb + (size_t)kt * DD;
#pragma unroll
            for (int p = 0; p < 4; ++p) {
                int line = t * 4 + p;   // 1024 lines x 64B = 64KB tile
                __builtin_prefetch(V0 + ((size_t)(line >> 5) * DD) + (line & 31) * 32, 0, 1);
            }
        }

        // ---- partial scores: S_w[16 q][32 key] over this wave's d-slice ----
        v8f sacc[2] = { vzero8(), vzero8() };
#pragma unroll
        for (int c = 0; c < 4; ++c) {
            AF kf0, kf1;
#pragma unroll
            for (int v = 0; v < 8; ++v) {
                int kb = wave * 128 + c * 32 + kB(v, g) + padk;
                kf0.u[v] = *(const unsigned int*)&Ks[(col)      * KS_STRIDE2 + kb];
                kf1.u[v] = *(const unsigned int*)&Ks[(16 + col) * KS_STRIDE2 + kb];
            }
            sacc[0] = __builtin_amdgcn_wmma_f32_16x16x32_bf16(
                false, qf[c].v, false, kf0.v, (short)0, sacc[0], false, false);
            sacc[1] = __builtin_amdgcn_wmma_f32_16x16x32_bf16(
                false, qf[c].v, false, kf1.v, (short)0, sacc[1], false, false);
        }
        {
            float* Sw = Sred + wave * 512;
#pragma unroll
            for (int nt = 0; nt < 2; ++nt)
#pragma unroll
                for (int i = 0; i < 8; ++i)
                    Sw[(i + 8 * g) * 32 + nt * 16 + col] = sacc[nt][i];
        }
        __syncthreads();

        // ---- cross-wave reduce + 1/sqrt(d) scale ----
        for (int idx = t; idx < 512; idx += 256) {
            float s = 0.0f;
#pragma unroll
            for (int w = 0; w < 8; ++w) s += Sred[w * 512 + idx];
            Sfin[idx] = s * 0.03125f;
        }
        __syncthreads();

        // ---- online softmax (one thread per query row) ----
        if (t < 16) {
            float m_old = row_m[t], mx = m_old;
#pragma unroll
            for (int j = 0; j < 32; ++j) mx = fmaxf(mx, Sfin[t * 32 + j]);
            float a = __expf(m_old - mx);
            float s = 0.0f;
#pragma unroll
            for (int j = 0; j < 32; ++j) {
                float p = __expf(Sfin[t * 32 + j] - mx);
                s += p;
                Ps[t * PS_STRIDE + j] = (bf16_t)p;
            }
            row_m[t] = mx;
            row_l[t] = row_l[t] * a + s;
            alpha[t] = a;
        }
        __syncthreads();

        // ---- rescale O accumulators ----
        float al[8];
#pragma unroll
        for (int i = 0; i < 8; ++i) al[i] = alpha[i + 8 * g];
#pragma unroll
        for (int nt = 0; nt < 8; ++nt)
#pragma unroll
            for (int i = 0; i < 8; ++i) of[nt][i] *= al[i];

        // ---- P fragment (A-matrix 16x32) ----
        AF pf;
#pragma unroll
        for (int v = 0; v < 8; ++v)
            pf.u[v] = *(const unsigned int*)&Ps[col * PS_STRIDE + kA(v, g)];

        // ---- stage V tile transposed: Vt[d][key] ----
        for (int vi = t; vi < 4096; vi += 256) {
            int key = vi >> 7;
            int d8  = (vi & 127) << 3;
            union { uint4 q; bf16_t h[8]; } cv;
            cv.q = *(const uint4*)&Vb[(size_t)(kt + key) * DD + d8];
#pragma unroll
            for (int j = 0; j < 8; ++j)
                Vt[(d8 + j) * VT_STRIDE + key] = cv.h[j];
        }
        __syncthreads();

        // ---- O += P x V (wave's 128-wide d-slice) ----
#pragma unroll
        for (int nt = 0; nt < 8; ++nt) {
            AF vf;
            int dcol = wave * 128 + nt * 16 + col;
#pragma unroll
            for (int v = 0; v < 8; ++v)
                vf.u[v] = *(const unsigned int*)&Vt[dcol * VT_STRIDE + kB(v, g)];
            of[nt] = __builtin_amdgcn_wmma_f32_16x16x32_bf16(
                false, pf.v, false, vf.v, (short)0, of[nt], false, false);
        }
    }

    // ---- normalize by row sums and write f32 output ----
    float rl[8];
#pragma unroll
    for (int i = 0; i < 8; ++i) rl[i] = 1.0f / row_l[i + 8 * g];
#pragma unroll
    for (int nt = 0; nt < 8; ++nt)
#pragma unroll
        for (int i = 0; i < 8; ++i)
            Ob[(size_t)(q0 + i + 8 * g) * DD + wave * 128 + nt * 16 + col] =
                of[nt][i] * rl[i];
}

// ---------------------------------------------------------------------------
extern "C" void kernel_launch(void* const* d_in, const int* in_sizes, int n_in,
                              void* d_out, int out_size, void* d_ws, size_t ws_size,
                              hipStream_t stream)
{
    const float* x  = (const float*)d_in[0];
    const float* Wq = (const float*)d_in[1];
    const float* bq = (const float*)d_in[2];
    const float* Wk = (const float*)d_in[3];
    const float* bk = (const float*)d_in[4];
    const float* Wv = (const float*)d_in[5];
    const float* bv = (const float*)d_in[6];
    float* out = (float*)d_out;

    const size_t ntok = (size_t)NB * SEQ * DD;   // 16,777,216 elems
    bf16_t* Qb  = (bf16_t*)d_ws;
    bf16_t* Kb  = Qb + ntok;
    bf16_t* Vb  = Kb + ntok;
    bf16_t* Xbf = Vb + ntok;                     // total 128 MB bf16 scratch

    cvt_bf16_kernel<<<dim3((unsigned)(ntok / (256 * 8))), 256, 0, stream>>>(x, Xbf);

    dim3 g1(DD / 128, (NB * SEQ) / 128, 3);      // 8 x 128 x 3
    qkv_gemm_kernel<<<g1, 256, 0, stream>>>(Xbf, Wq, bq, Wk, bk, Wv, bv,
                                            Qb, Kb, Vb);

    dim3 g2(SEQ / 16, NB);                       // 128 x 8
    flash_attn_kernel<<<g2, 256, SMEM_BYTES, stream>>>(Qb, Kb, Vb, out);
}
